// PPOActorRNN_65352222376256
// MI455X (gfx1250) — compile-verified, matrix operand order checked by reading, not run on hardware
//
#include <hip/hip_runtime.h>

// ---------------------------------------------------------------------------
// PPO Actor RNN forward for MI455X (gfx1250, wave32, WMMA).
// B=256, T=512, H=512, A=16, L=3.
// All GEMMs run on v_wmma_f32_16x16x32_bf16 with bf16 hi/lo splitting
// (x = hi + lo) accumulating hi*hi + lo*hi + hi*lo in f32 (~2^-17 rel err).
// Weight tiles are staged with GLOBAL_LOAD_ASYNC_TO_LDS_B128 (ASYNCcnt).
// ---------------------------------------------------------------------------

#define Bsz  256
#define Tsz  512
#define Hsz  512
#define Asz  16
#define BT   (Bsz * Tsz)        // 131072
#define TC   64                 // timestep chunk for gi precompute

typedef __attribute__((ext_vector_type(16))) __bf16        v16bf;
typedef __attribute__((ext_vector_type(8)))  float         v8f;
typedef __attribute__((ext_vector_type(8)))  unsigned int  v8u;

__device__ __forceinline__ v8f wmma_bf16(v8u a, v8u b, v8f c) {
  return __builtin_amdgcn_wmma_f32_16x16x32_bf16(
      false, __builtin_bit_cast(v16bf, a),
      false, __builtin_bit_cast(v16bf, b),
      (short)0, c, false, false);
}

// Async DMA: 16 bytes per lane, global memory -> LDS, tracked by ASYNCcnt.
// lds_off is the LDS byte offset (low 32 bits of the flat shared address:
// the LDS aperture occupies the high 32 bits per the CDNA5 aperture rules).
__device__ __forceinline__ void async_copy_b128(unsigned lds_off,
                                                const void* gaddr) {
  asm volatile("global_load_async_to_lds_b128 %0, %1, off"
               :
               : "v"(lds_off), "v"((unsigned long long)(uintptr_t)gaddr)
               : "memory");
}
__device__ __forceinline__ void wait_async() {
#if __has_builtin(__builtin_amdgcn_s_wait_asynccnt)
  __builtin_amdgcn_s_wait_asynccnt(0);
#else
  asm volatile("s_wait_asynccnt 0" ::: "memory");
#endif
}
__device__ __forceinline__ unsigned lds_offset_of(const void* shared_ptr) {
  return (unsigned)(uintptr_t)shared_ptr;  // low 32 bits = LDS byte offset
}

__device__ __forceinline__ unsigned short f2bf(float x) {
  unsigned int u = __builtin_bit_cast(unsigned int, x);
  u += 0x7FFFu + ((u >> 16) & 1u);          // round-to-nearest-even
  return (unsigned short)(u >> 16);
}
__device__ __forceinline__ float bf2f(unsigned short b) {
  unsigned int u = ((unsigned int)b) << 16;
  return __builtin_bit_cast(float, u);
}
// Split two f32 into packed bf16 hi-pair and lo-pair (k, k+1 in one dword).
__device__ __forceinline__ void split2(float x0, float x1,
                                       unsigned int& hi, unsigned int& lo) {
  unsigned short h0 = f2bf(x0), h1 = f2bf(x1);
  unsigned short l0 = f2bf(x0 - bf2f(h0)), l1 = f2bf(x1 - bf2f(h1));
  hi = (unsigned int)h0 | ((unsigned int)h1 << 16);
  lo = (unsigned int)l0 | ((unsigned int)l1 << 16);
}
// k2 (K-pair) index held by fragment VGPR v for this half-wave (ISA 16-bit A layout).
__device__ __forceinline__ int k2sel(int v, int hi4) {
  return ((v < 4) ? v : (v + 4)) + hi4;
}

// ---------------------------------------------------------------------------
// Pre-pack weight matrix W[N][K] (f32, row-major) into k-pair-interleaved
// bf16 hi/lo arrays of uint32, layout [K/2][N].
// ---------------------------------------------------------------------------
__global__ __launch_bounds__(256) void pack_w_kernel(
    const float* __restrict__ W, unsigned int* __restrict__ hi,
    unsigned int* __restrict__ lo, int N, int K) {
  int idx = blockIdx.x * 256 + threadIdx.x;
  int total = (K >> 1) * N;
  if (idx >= total) return;
  int k2 = idx / N, n = idx - k2 * N;
  float x0 = W[(size_t)n * K + 2 * k2];
  float x1 = W[(size_t)n * K + 2 * k2 + 1];
  unsigned int ph, pl;
  split2(x0, x1, ph, pl);
  hi[idx] = ph;
  lo[idx] = pl;
}

// ---------------------------------------------------------------------------
// Generic K=512 GEMM, WG tile = 32 rows x 512 cols, 256 threads (8 waves).
// GI=false : Y = relu(LN(X @ W^T + bias)) ; rows contiguous (m * 512).
// GI=true  : Y[m*Nw + ncb + n] = X @ W^T + bias ; rows mapped (t,b) for the
//            GRU input projection chunk (m = tl*256 + b, t = tbase + tl).
// Weights come pre-packed: Whi/Wlo are [256][Nw] uint32 (two bf16 per dword).
// ---------------------------------------------------------------------------
template <bool GI>
__global__ __launch_bounds__(256) void gemm512_kernel(
    const float* __restrict__ X, const unsigned int* __restrict__ Whi,
    const unsigned int* __restrict__ Wlo, const float* __restrict__ bias,
    const float* __restrict__ lng, const float* __restrict__ lnb,
    float* __restrict__ Y, int Nw, int tbase) {
  __shared__ __align__(16) unsigned int wt[16][512];   // 32KB: W chunk (hi, then lo); aliased as y16
  __shared__ __align__(16) unsigned int xhi[16][32];   // 2KB
  __shared__ __align__(16) unsigned int xlo[16][32];   // 2KB
  __shared__ float psum[16][16], psq[16][16];
  __shared__ float mu_s[16], rs_s[16];

  const int tid = threadIdx.x;
  const int lane = tid & 31, wv = tid >> 5;
  const int rw = wv >> 2, q = wv & 3;          // row-half (0/1), col-quarter
  const int lj = lane & 15;
  const int hi4 = (lane >= 16) ? 4 : 0;
  const int m0 = blockIdx.x * 32;
  const int ncb = blockIdx.y * 512;
  const unsigned wt_lds = lds_offset_of(&wt[0][0]);

  // staging role: row sm, k2-group skr
  const int sm = tid & 31, skr = tid >> 5;
  size_t xrow;
  {
    int mg = m0 + sm;
    if (GI) {
      int b = mg & 255;
      int tstep = tbase + (mg >> 8);
      xrow = ((size_t)b * Tsz + tstep) * Hsz;
    } else {
      xrow = (size_t)mg * Hsz;
    }
  }

  v8f c[8] = {};

  for (int kc = 0; kc < 16; ++kc) {
    const int kk2 = kc * 16;
    // ---- async-stage W-hi chunk (16 k2-rows x 512 cols) directly to LDS ----
#pragma unroll
    for (int rep = 0; rep < 8; ++rep) {
      int idx = rep * 256 + tid;
      int kr = idx >> 7, c4 = idx & 127;
      async_copy_b128(wt_lds + (unsigned)idx * 16,
                      Whi + (size_t)(kk2 + kr) * Nw + ncb + c4 * 4);
    }
    // ---- stage X split (hi and lo) through VALU (needs f32->bf16 split) ----
#pragma unroll
    for (int s = 0; s < 2; ++s) {
      int k2l = skr + s * 8;
      const float2 v = *(const float2*)(X + xrow + (size_t)(kk2 + k2l) * 2);
      unsigned int ph, pl;
      split2(v.x, v.y, ph, pl);
      xhi[k2l][sm] = ph;
      xlo[k2l][sm] = pl;
    }
    if (kc + 1 < 16)  // prefetch next W-hi chunk (global_prefetch_b8)
      __builtin_prefetch(Whi + (size_t)(kk2 + 16 + skr) * Nw + ncb + sm * 16, 0, 1);
    wait_async();
    __syncthreads();

    v8u ahi, alo;
#pragma unroll
    for (int v = 0; v < 8; ++v) {
      int k2b = k2sel(v, hi4);
      ahi[v] = xhi[k2b][16 * rw + lj];
      alo[v] = xlo[k2b][16 * rw + lj];
    }
#pragma unroll
    for (int t = 0; t < 8; ++t) {
      int col = q * 128 + t * 16 + lj;
      v8u b;
#pragma unroll
      for (int v = 0; v < 8; ++v) b[v] = wt[k2sel(v, hi4)][col];
      c[t] = wmma_bf16(ahi, b, c[t]);
      c[t] = wmma_bf16(alo, b, c[t]);
    }
    __syncthreads();
    // ---- async-restage W-lo into the same LDS buffer ----
#pragma unroll
    for (int rep = 0; rep < 8; ++rep) {
      int idx = rep * 256 + tid;
      int kr = idx >> 7, c4 = idx & 127;
      async_copy_b128(wt_lds + (unsigned)idx * 16,
                      Wlo + (size_t)(kk2 + kr) * Nw + ncb + c4 * 4);
    }
    wait_async();
    __syncthreads();
#pragma unroll
    for (int t = 0; t < 8; ++t) {
      int col = q * 128 + t * 16 + lj;
      v8u b;
#pragma unroll
      for (int v = 0; v < 8; ++v) b[v] = wt[k2sel(v, hi4)][col];
      c[t] = wmma_bf16(ahi, b, c[t]);
    }
    __syncthreads();
  }

  if (GI) {
    // bias + direct store
#pragma unroll
    for (int t = 0; t < 8; ++t) {
      int nl = q * 128 + t * 16 + lj;
      float bb = bias[ncb + nl];
#pragma unroll
      for (int r = 0; r < 8; ++r) {
        int ml = 16 * rw + (hi4 ? 8 : 0) + r;
        Y[(size_t)(m0 + ml) * Nw + ncb + nl] = c[t][r] + bb;
      }
    }
  } else {
    // fused bias + LayerNorm + ReLU, 16 rows per phase (reuses wt as f32 y16)
    float* y16 = (float*)&wt[0][0];
    for (int ph = 0; ph < 2; ++ph) {
      if (rw == ph) {
#pragma unroll
        for (int t = 0; t < 8; ++t) {
          int nl = q * 128 + t * 16 + lj;
          float bb = bias[nl];
#pragma unroll
          for (int r = 0; r < 8; ++r) {
            int ml = (hi4 ? 8 : 0) + r;
            y16[ml * 512 + nl] = c[t][r] + bb;
          }
        }
      }
      __syncthreads();
      int row = tid >> 4, cg = tid & 15;
      const float* yr = y16 + row * 512 + cg * 32;
      float s = 0.f, s2 = 0.f;
#pragma unroll
      for (int i = 0; i < 32; ++i) {
        float v = yr[i];
        s += v;
        s2 += v * v;
      }
      psum[row][cg] = s;
      psq[row][cg] = s2;
      __syncthreads();
      if (tid < 16) {
        float S = 0.f, S2 = 0.f;
#pragma unroll
        for (int j = 0; j < 16; ++j) {
          S += psum[tid][j];
          S2 += psq[tid][j];
        }
        float mu = S * (1.0f / 512.0f);
        float var = S2 * (1.0f / 512.0f) - mu * mu;
        mu_s[tid] = mu;
        rs_s[tid] = rsqrtf(var + 1e-5f);
      }
      __syncthreads();
      float mu = mu_s[row], rs = rs_s[row];
      float* yo = Y + (size_t)(m0 + ph * 16 + row) * 512 + cg * 32;
#pragma unroll
      for (int i = 0; i < 32; ++i) {
        int col = cg * 32 + i;
        float v = (yr[i] - mu) * rs * lng[col] + lnb[col];
        yo[i] = fmaxf(v, 0.0f);
      }
      __syncthreads();
    }
  }
}

// ---------------------------------------------------------------------------
// One GRU timestep, fused: gh = h_masked @ Whh^T (+bhh) for all 3 gates of a
// 128-column slice, then gate math + state update.
// Grid (8, 4): 32 rows(b) x 128 h-cols per WG; wave handles 2 tiles per gate.
// ---------------------------------------------------------------------------
__global__ __launch_bounds__(256) void gru_step_kernel(
    const float* __restrict__ h_in, float* __restrict__ h_out,
    const float* __restrict__ gi,                  // [TC*B, 1536] chunk
    const unsigned int* __restrict__ Whi, const unsigned int* __restrict__ Wlo,
    const float* __restrict__ bhh, const unsigned char* __restrict__ dones,
    float* __restrict__ outs, int tl, int t) {
  __shared__ __align__(16) unsigned int wt[16][384];   // 24KB (hi then lo)
  __shared__ __align__(16) unsigned int xhi2[16][32];
  __shared__ __align__(16) unsigned int xlo2[16][32];

  const int tid = threadIdx.x;
  const int lane = tid & 31, wv = tid >> 5;
  const int rw = wv >> 2, q = wv & 3;
  const int lj = lane & 15;
  const int hi4 = (lane >= 16) ? 4 : 0;
  const int b0 = blockIdx.x * 32;
  const int n0 = blockIdx.y * 128;
  const unsigned wt_lds = lds_offset_of(&wt[0][0]);

  const int sm = tid & 31, skr = tid >> 5;
  const float smask = dones[(size_t)(b0 + sm) * Tsz + t] ? 0.f : 1.f;
  const float* hrow = h_in + (size_t)(b0 + sm) * Hsz;

  v8f c[6] = {};

  for (int kc = 0; kc < 16; ++kc) {
    const int kk2 = kc * 16;
    // async-stage Whh-hi: 16 k2-rows x (3 gates x 128 cols)
#pragma unroll
    for (int rep = 0; rep < 6; ++rep) {
      int idx = rep * 256 + tid;
      int kr = idx / 96, cc = idx - kr * 96;
      int col4 = cc * 4;
      int gate = col4 >> 7, j = col4 & 127;
      async_copy_b128(wt_lds + (unsigned)idx * 16,
                      Whi + (size_t)(kk2 + kr) * 1536 + gate * 512 + n0 + j);
    }
    // stage masked h split through VALU
#pragma unroll
    for (int s = 0; s < 2; ++s) {
      int k2l = skr + s * 8;
      float2 v = *(const float2*)(hrow + (size_t)(kk2 + k2l) * 2);
      v.x *= smask;
      v.y *= smask;
      unsigned int ph, pl;
      split2(v.x, v.y, ph, pl);
      xhi2[k2l][sm] = ph;
      xlo2[k2l][sm] = pl;
    }
    wait_async();
    __syncthreads();

    v8u ahi, alo;
#pragma unroll
    for (int v = 0; v < 8; ++v) {
      int k2b = k2sel(v, hi4);
      ahi[v] = xhi2[k2b][16 * rw + lj];
      alo[v] = xlo2[k2b][16 * rw + lj];
    }
#pragma unroll
    for (int tt = 0; tt < 6; ++tt) {
      int col = (tt >> 1) * 128 + q * 32 + (tt & 1) * 16 + lj;
      v8u b;
#pragma unroll
      for (int v = 0; v < 8; ++v) b[v] = wt[k2sel(v, hi4)][col];
      c[tt] = wmma_bf16(ahi, b, c[tt]);
      c[tt] = wmma_bf16(alo, b, c[tt]);
    }
    __syncthreads();
    // async-restage Whh-lo
#pragma unroll
    for (int rep = 0; rep < 6; ++rep) {
      int idx = rep * 256 + tid;
      int kr = idx / 96, cc = idx - kr * 96;
      int col4 = cc * 4;
      int gate = col4 >> 7, j = col4 & 127;
      async_copy_b128(wt_lds + (unsigned)idx * 16,
                      Wlo + (size_t)(kk2 + kr) * 1536 + gate * 512 + n0 + j);
    }
    wait_async();
    __syncthreads();
#pragma unroll
    for (int tt = 0; tt < 6; ++tt) {
      int col = (tt >> 1) * 128 + q * 32 + (tt & 1) * 16 + lj;
      v8u b;
#pragma unroll
      for (int v = 0; v < 8; ++v) b[v] = wt[k2sel(v, hi4)][col];
      c[tt] = wmma_bf16(ahi, b, c[tt]);
    }
    __syncthreads();
  }

  // ---- fused gate math + state update ----
#pragma unroll
  for (int t2 = 0; t2 < 2; ++t2) {
    int nh = n0 + q * 32 + t2 * 16 + lj;
    float br = bhh[nh], bz = bhh[512 + nh], bn = bhh[1024 + nh];
#pragma unroll
    for (int r = 0; r < 8; ++r) {
      int ml = 16 * rw + (hi4 ? 8 : 0) + r;
      int b = b0 + ml;
      const float* gir = gi + ((size_t)tl * Bsz + b) * 1536;
      float ir = gir[nh], iz = gir[512 + nh], inn = gir[1024 + nh];
      float hr = c[0 + t2][r] + br;
      float hz = c[2 + t2][r] + bz;
      float hn = c[4 + t2][r] + bn;
      float rg = 1.0f / (1.0f + expf(-(ir + hr)));
      float z = 1.0f / (1.0f + expf(-(iz + hz)));
      float nn = tanhf(inn + rg * hn);
      float mk = dones[(size_t)b * Tsz + t] ? 0.f : 1.f;
      float hp = h_in[(size_t)b * Hsz + nh] * mk;
      float hnew = (1.0f - z) * nn + z * hp;
      h_out[(size_t)b * Hsz + nh] = hnew;
      outs[((size_t)b * Tsz + t) * Hsz + nh] = hnew;
    }
  }
}

// ---------------------------------------------------------------------------
// Actor head projection to A=16 logits + avail-action masking (tiny GEMM).
// ---------------------------------------------------------------------------
__global__ __launch_bounds__(256) void head_logits_kernel(
    const float* __restrict__ am, const float* __restrict__ w2,
    const float* __restrict__ b2, const float* __restrict__ avail,
    float* __restrict__ out) {
  __shared__ float ws2[512][16];   // w2 transposed: [k][a]
  int tid = threadIdx.x;
#pragma unroll
  for (int rep = 0; rep < 32; ++rep) {
    int idx = rep * 256 + tid;     // a*512 + k
    int a0 = idx >> 9, k0 = idx & 511;
    ws2[k0][a0] = w2[idx];
  }
  __syncthreads();
  int mr = tid >> 4, a = tid & 15;
  size_t m = (size_t)blockIdx.x * 16 + mr;
  const float* xr = am + m * 512;
  float s = 0.f;
  for (int k = 0; k < 512; k += 4) {
    float4 x = *(const float4*)(xr + k);
    s += x.x * ws2[k][a] + x.y * ws2[k + 1][a] + x.z * ws2[k + 2][a] +
         x.w * ws2[k + 3][a];
  }
  s += b2[a];
  float av = avail[m * 16 + a];
  out[m * 16 + a] = s - (1.0f - av) * 1e10f;
}

// ---------------------------------------------------------------------------
// Host orchestration.
// ---------------------------------------------------------------------------
extern "C" void kernel_launch(void* const* d_in, const int* in_sizes, int n_in,
                              void* d_out, int out_size, void* d_ws,
                              size_t ws_size, hipStream_t stream) {
  (void)in_sizes; (void)n_in; (void)out_size; (void)ws_size;

  const float* hidden        = (const float*)d_in[0];
  const float* obs           = (const float*)d_in[1];
  const unsigned char* dones = (const unsigned char*)d_in[2];
  const float* avail         = (const float*)d_in[3];
  const float* in_w          = (const float*)d_in[4];
  const float* in_b          = (const float*)d_in[5];
  const float* ln_g          = (const float*)d_in[6];
  const float* ln_b          = (const float*)d_in[7];
  const float* wih           = (const float*)d_in[8];
  const float* whh           = (const float*)d_in[9];
  const float* bih           = (const float*)d_in[10];
  const float* bhh           = (const float*)d_in[11];
  const float* am1w          = (const float*)d_in[12];
  const float* am1b          = (const float*)d_in[13];
  const float* amlng         = (const float*)d_in[14];
  const float* amlnb         = (const float*)d_in[15];
  const float* am2w          = (const float*)d_in[16];
  const float* am2b          = (const float*)d_in[17];

  float* out_h = (float*)d_out;                   // [B,H]
  float* out_logits = out_h + (size_t)Bsz * Hsz;  // [B,T,A]

  // ---- workspace carve ----
  char* p = (char*)d_ws;
  auto carve = [&](size_t bytes) -> char* {
    char* r = p;
    p += (bytes + 255) & ~(size_t)255;
    return r;
  };
  const size_t BTH = (size_t)BT * Hsz;
  float* bufA = (float*)carve(BTH * 4);
  float* bufB = (float*)carve(BTH * 4);
  float* giC  = (float*)carve((size_t)TC * Bsz * 1536 * 4);
  float* h0   = (float*)carve((size_t)Bsz * Hsz * 4);
  float* h1   = (float*)carve((size_t)Bsz * Hsz * 4);
  unsigned int* inw_hi = (unsigned int*)carve((size_t)3 * 131072 * 4);
  unsigned int* inw_lo = (unsigned int*)carve((size_t)3 * 131072 * 4);
  unsigned int* wih_hi = (unsigned int*)carve((size_t)393216 * 4);
  unsigned int* wih_lo = (unsigned int*)carve((size_t)393216 * 4);
  unsigned int* whh_hi = (unsigned int*)carve((size_t)393216 * 4);
  unsigned int* whh_lo = (unsigned int*)carve((size_t)393216 * 4);
  unsigned int* am1_hi = (unsigned int*)carve((size_t)131072 * 4);
  unsigned int* am1_lo = (unsigned int*)carve((size_t)131072 * 4);

  // ---- pre-pack weights to bf16 hi/lo, [K/2][N] k-pair-interleaved ----
  for (int l = 0; l < 3; ++l)
    pack_w_kernel<<<512, 256, 0, stream>>>(in_w + (size_t)l * 262144,
                                           inw_hi + (size_t)l * 131072,
                                           inw_lo + (size_t)l * 131072, 512, 512);
  pack_w_kernel<<<1536, 256, 0, stream>>>(wih, wih_hi, wih_lo, 1536, 512);
  pack_w_kernel<<<1536, 256, 0, stream>>>(whh, whh_hi, whh_lo, 1536, 512);
  pack_w_kernel<<<512, 256, 0, stream>>>(am1w, am1_hi, am1_lo, 512, 512);

  // ---- pre-RNN MLP: 3 x (Linear -> LN -> ReLU) ----
  gemm512_kernel<false><<<dim3(4096, 1), 256, 0, stream>>>(
      obs, inw_hi, inw_lo, in_b, ln_g, ln_b, bufA, 512, 0);
  gemm512_kernel<false><<<dim3(4096, 1), 256, 0, stream>>>(
      bufA, inw_hi + 131072, inw_lo + 131072, in_b + 512, ln_g + 512,
      ln_b + 512, bufB, 512, 0);
  gemm512_kernel<false><<<dim3(4096, 1), 256, 0, stream>>>(
      bufB, inw_hi + 262144, inw_lo + 262144, in_b + 1024, ln_g + 1024,
      ln_b + 1024, bufA, 512, 0);

  hipMemcpyAsync(h0, hidden, (size_t)Bsz * Hsz * 4, hipMemcpyDeviceToDevice,
                 stream);

  // ---- GRU scan: gi precomputed per 64-step chunk, then sequential steps ----
  float* hin = h0;
  float* hout = h1;
  for (int cchunk = 0; cchunk < Tsz / TC; ++cchunk) {
    gemm512_kernel<true><<<dim3((TC * Bsz) / 32, 3), 256, 0, stream>>>(
        bufA, wih_hi, wih_lo, bih, nullptr, nullptr, giC, 1536, cchunk * TC);
    for (int tl = 0; tl < TC; ++tl) {
      int t = cchunk * TC + tl;
      gru_step_kernel<<<dim3(8, 4), 256, 0, stream>>>(
          hin, hout, giC, whh_hi, whh_lo, bhh, dones, bufB, tl, t);
      float* tmp = hin;
      hin = hout;
      hout = tmp;
    }
  }
  // 512 swaps (even) -> final hidden state is back in h0 == hin
  hipMemcpyAsync(out_h, hin, (size_t)Bsz * Hsz * 4, hipMemcpyDeviceToDevice,
                 stream);

  // ---- actor head ----
  gemm512_kernel<false><<<dim3(4096, 1), 256, 0, stream>>>(
      bufB, am1_hi, am1_lo, am1b, amlng, amlnb, bufA, 512, 0);
  head_logits_kernel<<<BT / 16, 256, 0, stream>>>(bufA, am2w, am2b, avail,
                                                  out_logits);
}
